// CausalSelfAttention_17592186045217
// MI455X (gfx1250) — matrix operand order repaired and checked by reading, not compiled, not run on hardware
//
#include <hip/hip_runtime.h>

typedef __bf16 bf16_t;
typedef __attribute__((ext_vector_type(8)))  __bf16 v8bf;
typedef __attribute__((ext_vector_type(16))) __bf16 v16bf;
typedef __attribute__((ext_vector_type(8)))  float  v8f;

#define B_    2
#define S_    2048
#define DIM_  2048
#define H_    16
#define HKV_  4
#define D_    128
#define KVD_  512
#define GROUP_ 4

static __device__ __forceinline__ v16bf make_afrag(v8bf a0, v8bf a1) {
  // A-fragment (16x32 bf16): elements 0..7 = a0, 8..15 = a1 (k order set by the caller's offsets)
  return __builtin_shufflevector(a0, a1, 0,1,2,3,4,5,6,7,8,9,10,11,12,13,14,15);
}

// ---------------------------------------------------------------- f32 -> bf16
__global__ void f32_to_bf16_kernel(const float* __restrict__ src,
                                   bf16_t* __restrict__ dst, long n) {
  long i = (long)blockIdx.x * blockDim.x + threadIdx.x;
  if (i < n) dst[i] = (bf16_t)src[i];
}

// ------------------------------------------------- C = A(MxK) @ B(NxK)^T, f32 out
// Block = 8 waves = 256x64 tile. All waves share one 64-column B stripe, staged
// in LDS by ASYNCcnt-tracked global_load_async_to_lds_b128, double-buffered so
// the next 4KB B tile streams in while 8 WMMAs/wave run on the current one.
// Per wave per k-step: 4 b128 A loads + 8 ds b128 B loads -> 8 WMMAs.
__global__ __launch_bounds__(256)
void gemm_bf16_nt_wmma(const bf16_t* __restrict__ A, const bf16_t* __restrict__ Bw,
                       float* __restrict__ C, int M, int N, int K) {
  __shared__ bf16_t bsh[2][64 * 32];            // 2 x 4KB B tiles (64 rows x 32 k)

  int t    = threadIdx.x;
  int w    = t >> 5;                            // wave in block (0..7)
  int lane = t & 31;
  int hi = lane >> 4, lo = lane & 15;

  int tilesN = N >> 6;
  int bm = (int)blockIdx.x / tilesN;            // 256-row block tile
  int bn = (int)blockIdx.x - bm * tilesN;       // 64-col block tile
  if (bm * 256 >= M) return;

  // --- cooperative async B copy: thread t moves 16B (row t>>2, cols (t&3)*8)
  int brow = t >> 2;                            // 0..63
  int bcol = (t & 3) * 8;                       // 0,8,16,24 (bf16 elems)
  const bf16_t* gB = Bw + (size_t)(bn * 64 + brow) * K + bcol;
  unsigned ldsoff = (unsigned)((brow * 32 + bcol) * 2);
  unsigned lbase[2] = { (unsigned)(size_t)&bsh[0][0] + ldsoff,
                        (unsigned)(size_t)&bsh[1][0] + ldsoff };

  // --- per-wave A row pointers (rows bm*256 + w*32 ...)
  const bf16_t* ap0 = A + (size_t)(bm * 256 + w * 32 + lo) * K + hi * 8;
  const bf16_t* ap1 = ap0 + (size_t)16 * K;

  v8f acc[2][4];
#pragma unroll
  for (int i = 0; i < 2; ++i)
#pragma unroll
    for (int j = 0; j < 4; ++j) acc[i][j] = v8f{};

  // prologue: stage k=0 tile into buffer 0
  {
    unsigned long long ga = (unsigned long long)gB;
    asm volatile("global_load_async_to_lds_b128 %0, %1, off"
                 :: "v"(lbase[0]), "v"(ga) : "memory");
  }

  for (int k = 0; k < K; k += 32) {
    int buf = (k >> 5) & 1;
    // issue next tile into the other buffer, then wait for the *previous* copy
    if (k + 32 < K) {
      unsigned long long ga = (unsigned long long)(gB + k + 32);
      asm volatile("global_load_async_to_lds_b128 %0, %1, off"
                   :: "v"(lbase[buf ^ 1]), "v"(ga) : "memory");
      asm volatile("s_wait_asynccnt 0x1" ::: "memory");
    } else {
      asm volatile("s_wait_asynccnt 0x0" ::: "memory");
    }
    __syncthreads();                            // whole 64x32 tile now visible

    v16bf a0 = make_afrag(*(const v8bf*)(ap0 + k), *(const v8bf*)(ap0 + k + 16));
    v16bf a1 = make_afrag(*(const v8bf*)(ap1 + k), *(const v8bf*)(ap1 + k + 16));
#pragma unroll
    for (int j = 0; j < 4; ++j) {
      // B fragment from LDS: lane = column j*16+lo, elements 0..7 = k hi*8.., 8..15 = k 16+hi*8..
      v16bf bfr = make_afrag(*(const v8bf*)&bsh[buf][(j * 16 + lo) * 32 + hi * 8], *(const v8bf*)&bsh[buf][(j * 16 + lo) * 32 + hi * 8 + 16]);
      acc[0][j] = __builtin_amdgcn_wmma_f32_16x16x32_bf16(false, a0, false, bfr,
                                                          (short)0, acc[0][j], false, false);
      acc[1][j] = __builtin_amdgcn_wmma_f32_16x16x32_bf16(false, a1, false, bfr,
                                                          (short)0, acc[1][j], false, false);
    }
    __syncthreads();                            // all waves done with buf before reuse
  }

  // C/D layout: VGPR r -> row r + 8*hi, column lo
#pragma unroll
  for (int i = 0; i < 2; ++i) {
#pragma unroll
    for (int j = 0; j < 4; ++j) {
      float* cp = C + (size_t)(bm * 256 + w * 32 + i * 16 + 8 * hi) * N
                    + bn * 64 + j * 16 + lo;
#pragma unroll
      for (int r = 0; r < 8; ++r) cp[(size_t)r * N] = acc[i][j][r];
    }
  }
}

// ---------------------------------------- per-(token, head) RMSNorm + RoPE (+gain)
__global__ void rmsnorm_rope_kernel(const float* __restrict__ t,
                                    bf16_t* __restrict__ out,
                                    const float* __restrict__ gain, int nh) {
  int wid  = (int)((blockIdx.x * blockDim.x + threadIdx.x) >> 5);
  int lane = threadIdx.x & 31;
  int h   = wid % nh;
  int tok = wid / nh;
  if (tok >= B_ * S_) return;
  int spos = tok % S_;
  const float* row = t + (size_t)tok * (nh * D_) + (size_t)h * D_;

  float v0 = row[lane], v1 = row[lane + 32], v2 = row[lane + 64], v3 = row[lane + 96];
  float ss = v0 * v0 + v1 * v1 + v2 * v2 + v3 * v3;
#pragma unroll
  for (int off = 16; off >= 1; off >>= 1) ss += __shfl_xor(ss, off, 32);
  float inv = rsqrtf(ss * (1.0f / 128.0f) + 1.1920929e-07f);
  v0 *= inv; v1 *= inv; v2 *= inv; v3 *= inv;

  const float c = 0.14391156831212787f;               // ln(10000)/64
  float a0 = (float)spos * __expf(-(float)lane * c);
  float a1 = (float)spos * __expf(-(float)(lane + 32) * c);
  float c0 = cosf(a0), s0 = sinf(a0), c1 = cosf(a1), s1 = sinf(a1);
  float o0 =  v0 * c0 + v2 * s0;
  float o2 = -v0 * s0 + v2 * c0;
  float o1 =  v1 * c1 + v3 * s1;
  float o3 = -v1 * s1 + v3 * c1;

  float g = gain ? gain[h] : 1.0f;
  bf16_t* orow = out + (size_t)tok * (nh * D_) + (size_t)h * D_;
  orow[lane]      = (bf16_t)(o0 * g);
  orow[lane + 32] = (bf16_t)(o1 * g);
  orow[lane + 64] = (bf16_t)(o2 * g);
  orow[lane + 96] = (bf16_t)(o3 * g);
}

// --------------------------- V (tok, KVD) f32 -> Vt[b][hkv][d][s] bf16 (transposed)
__global__ void v_transpose_bf16_kernel(const float* __restrict__ V,
                                        bf16_t* __restrict__ vt) {
  long i = (long)blockIdx.x * blockDim.x + threadIdx.x;
  long n = (long)B_ * S_ * KVD_;
  if (i >= n) return;
  int tok = (int)(i / KVD_);
  int c   = (int)(i % KVD_);
  int hk = c / D_, d = c % D_;
  int b = tok / S_, s = tok % S_;
  vt[(((size_t)b * HKV_ + hk) * D_ + d) * S_ + s] = (bf16_t)V[i];
}

// -------------------------------------------------------------- flash attention
// One wave per (b, h, 16-row q tile). 32-key blocks, online softmax.
__global__ __launch_bounds__(256)
void flash_attn_kernel(const bf16_t* __restrict__ qb, const bf16_t* __restrict__ kb,
                       const bf16_t* __restrict__ vt, float* __restrict__ y) {
  __shared__ bf16_t pbuf[8][16 * 32];   // 1 KB per wave P staging
  int widb = threadIdx.x >> 5;
  int wid  = (int)((blockIdx.x * blockDim.x + threadIdx.x) >> 5);
  int lane = threadIdx.x & 31;
  int hi = lane >> 4, lo = lane & 15;

  const int QT = S_ / 16;               // 128 q tiles
  int qt = wid % QT;
  int h  = (wid / QT) % H_;
  int b  = wid / (QT * H_);
  if (b >= B_) return;
  int hk = h / GROUP_;

  // Q fragments for all 4 K-steps of D=128 (A layout), loaded once.
  v16bf qf[4];
  {
    const bf16_t* qp = qb + (size_t)(b * S_ + qt * 16 + lo) * DIM_ + h * D_ + hi * 16;
#pragma unroll
    for (int ks = 0; ks < 4; ++ks) {
      v8bf a0 = *(const v8bf*)(qp + ks * 32);
      v8bf a1 = *(const v8bf*)(qp + ks * 32 + 8);
      qf[ks] = make_afrag(a0, a1);
    }
  }

  v8f o[8];
#pragma unroll
  for (int i = 0; i < 8; ++i) o[i] = v8f{};
  float m[8], l[8];
#pragma unroll
  for (int r = 0; r < 8; ++r) { m[r] = -1e30f; l[r] = 0.0f; }

  const float scale = 0.088388347648318447f;   // 1/sqrt(128)
  int nkeys = qt * 16 + 16;
  int kend  = (nkeys + 31) & ~31;

  const bf16_t* kbase = kb + (size_t)b * S_ * KVD_ + (size_t)hk * D_;
  const bf16_t* vbase = vt + (((size_t)b * HKV_ + hk) * D_) * S_;

  for (int kblk = 0; kblk < kend; kblk += 32) {
    // ---- scores: two 16x16 tiles (keys kblk..+15, kblk+16..+31)
    v8f s0 = {}, s1 = {};
    const bf16_t* kp0 = kbase + (size_t)(kblk + lo) * KVD_ + hi * 16;
    const bf16_t* kp1 = kbase + (size_t)(kblk + 16 + lo) * KVD_ + hi * 16;
#pragma unroll
    for (int ks = 0; ks < 4; ++ks) {
      v16bf k0 = *(const v16bf*)(kp0 + ks * 32);
      v16bf k1 = *(const v16bf*)(kp1 + ks * 32);
      s0 = __builtin_amdgcn_wmma_f32_16x16x32_bf16(false, qf[ks], false, k0,
                                                   (short)0, s0, false, false);
      s1 = __builtin_amdgcn_wmma_f32_16x16x32_bf16(false, qf[ks], false, k1,
                                                   (short)0, s1, false, false);
    }
    // ---- causal mask + online softmax (row = r + 8*hi, col = lo within tile)
    float corr[8];
#pragma unroll
    for (int r = 0; r < 8; ++r) {
      int qidx = qt * 16 + r + 8 * hi;
      float a = (kblk + lo      <= qidx) ? s0[r] * scale : -1e30f;
      float c = (kblk + 16 + lo <= qidx) ? s1[r] * scale : -1e30f;
      float mx = fmaxf(a, c);
#pragma unroll
      for (int off = 1; off < 16; off <<= 1) mx = fmaxf(mx, __shfl_xor(mx, off, 32));
      float mn = fmaxf(m[r], mx);
      float cf = __expf(m[r] - mn);
      m[r] = mn;
      float p0 = __expf(a - mn);
      float p1 = __expf(c - mn);
      float rs = p0 + p1;
#pragma unroll
      for (int off = 1; off < 16; off <<= 1) rs += __shfl_xor(rs, off, 32);
      l[r] = l[r] * cf + rs;
      corr[r] = cf;
      s0[r] = p0; s1[r] = p1;
    }
#pragma unroll
    for (int dc = 0; dc < 8; ++dc)
#pragma unroll
      for (int r = 0; r < 8; ++r) o[dc][r] *= corr[r];

    // ---- P: C-layout -> LDS (row-major 16x32) -> A-layout fragment
    bf16_t* pb = &pbuf[widb][0];
#pragma unroll
    for (int r = 0; r < 8; ++r) {
      pb[(r + 8 * hi) * 32 + lo]      = (bf16_t)s0[r];
      pb[(r + 8 * hi) * 32 + 16 + lo] = (bf16_t)s1[r];
    }
    __builtin_amdgcn_wave_barrier();   // DS in-order per wave; block compiler reorder
    v8bf p0 = *(const v8bf*)(pb + lo * 32 + hi * 8);
    v8bf p1 = *(const v8bf*)(pb + lo * 32 + hi * 8 + 16);
    v16bf pf = make_afrag(p0, p1);
    __builtin_amdgcn_wave_barrier();

    // ---- O += P (16x32) @ V (32x16 per d-chunk); Vt gives contiguous key runs
#pragma unroll
    for (int dc = 0; dc < 8; ++dc) {
      v16bf vf = make_afrag(*(const v8bf*)(vbase + (size_t)(dc * 16 + lo) * S_ + kblk + hi * 8), *(const v8bf*)(vbase + (size_t)(dc * 16 + lo) * S_ + kblk + hi * 8 + 16));
      o[dc] = __builtin_amdgcn_wmma_f32_16x16x32_bf16(false, pf, false, vf,
                                                      (short)0, o[dc], false, false);
    }
  }

  // ---- normalize by l and write y (tok, DIM) f32
  float* yp = y + (size_t)(b * S_ + qt * 16 + 8 * hi) * DIM_ + h * D_ + lo;
#pragma unroll
  for (int r = 0; r < 8; ++r) {
    float invl = 1.0f / l[r];
#pragma unroll
    for (int dc = 0; dc < 8; ++dc)
      yp[(size_t)r * DIM_ + dc * 16] = o[dc][r] * invl;
  }
}

// ------------------------- y - (y . vhat) vhat per (token, kv-head, group head)
__global__ void vproj_ortho_kernel(const float* __restrict__ y,
                                   const float* __restrict__ V,
                                   bf16_t* __restrict__ ybf) {
  int wid  = (int)((blockIdx.x * blockDim.x + threadIdx.x) >> 5);
  int lane = threadIdx.x & 31;
  int hk  = wid % HKV_;
  int tok = wid / HKV_;
  if (tok >= B_ * S_) return;

  const float* vp = V + (size_t)tok * KVD_ + (size_t)hk * D_;
  float vv[4];
  float ss = 0.0f;
#pragma unroll
  for (int j = 0; j < 4; ++j) { vv[j] = vp[lane + 32 * j]; ss += vv[j] * vv[j]; }
#pragma unroll
  for (int off = 16; off >= 1; off >>= 1) ss += __shfl_xor(ss, off, 32);
  float inv = 1.0f / fmaxf(sqrtf(ss), 1e-12f);
#pragma unroll
  for (int j = 0; j < 4; ++j) vv[j] *= inv;

#pragma unroll
  for (int g = 0; g < GROUP_; ++g) {
    int h = hk * GROUP_ + g;
    const float* yp = y + (size_t)tok * DIM_ + (size_t)h * D_;
    float yy[4];
    float dot = 0.0f;
#pragma unroll
    for (int j = 0; j < 4; ++j) { yy[j] = yp[lane + 32 * j]; dot += yy[j] * vv[j]; }
#pragma unroll
    for (int off = 16; off >= 1; off >>= 1) dot += __shfl_xor(dot, off, 32);
    bf16_t* op = ybf + (size_t)tok * DIM_ + (size_t)h * D_;
#pragma unroll
    for (int j = 0; j < 4; ++j) op[lane + 32 * j] = (bf16_t)(yy[j] - dot * vv[j]);
  }
}

// ================================================================= launcher
extern "C" void kernel_launch(void* const* d_in, const int* in_sizes, int n_in,
                              void* d_out, int out_size, void* d_ws, size_t ws_size,
                              hipStream_t stream) {
  (void)in_sizes; (void)n_in; (void)out_size; (void)ws_size;
  const float* x  = (const float*)d_in[0];
  const float* Wq = (const float*)d_in[1];
  const float* Wk = (const float*)d_in[2];
  const float* Wv = (const float*)d_in[3];
  const float* Wo = (const float*)d_in[4];
  const float* qg = (const float*)d_in[5];
  float* out = (float*)d_out;

  const long TOK = (long)B_ * S_;                  // 4096
  char* ws = (char*)d_ws;
  auto carve = [&](size_t bytes) -> char* {
    char* p = ws;
    ws += (bytes + 255) & ~(size_t)255;
    return p;
  };
  bf16_t* xb  = (bf16_t*)carve((size_t)TOK * DIM_ * 2);
  bf16_t* wqb = (bf16_t*)carve((size_t)DIM_ * DIM_ * 2);
  bf16_t* wkb = (bf16_t*)carve((size_t)KVD_ * DIM_ * 2);
  bf16_t* wvb = (bf16_t*)carve((size_t)KVD_ * DIM_ * 2);
  bf16_t* wob = (bf16_t*)carve((size_t)DIM_ * DIM_ * 2);
  float*  Qf  = (float*)carve((size_t)TOK * DIM_ * 4);
  float*  Kf  = (float*)carve((size_t)TOK * KVD_ * 4);
  float*  Vf  = (float*)carve((size_t)TOK * KVD_ * 4);
  bf16_t* qbb = (bf16_t*)carve((size_t)TOK * DIM_ * 2);
  bf16_t* kbb = (bf16_t*)carve((size_t)TOK * KVD_ * 2);
  bf16_t* vtb = (bf16_t*)carve((size_t)TOK * KVD_ * 2);
  float*  yf  = (float*)carve((size_t)TOK * DIM_ * 4);
  bf16_t* ybf = (bf16_t*)carve((size_t)TOK * DIM_ * 2);

  dim3 blk(256);
  auto cvt = [&](const float* s, bf16_t* d, long n) {
    f32_to_bf16_kernel<<<dim3((unsigned)((n + 255) / 256)), blk, 0, stream>>>(s, d, n);
  };

  // 1) downconvert activations + weights to bf16
  cvt(x,  xb,  TOK * DIM_);
  cvt(Wq, wqb, (long)DIM_ * DIM_);
  cvt(Wk, wkb, (long)KVD_ * DIM_);
  cvt(Wv, wvb, (long)KVD_ * DIM_);
  cvt(Wo, wob, (long)DIM_ * DIM_);

  // 2) QKV projections (WMMA GEMMs, async-LDS B staging): X(4096x2048) @ W^T
  {
    int blocksQ = (int)(TOK / 256) * (DIM_ / 64);   // 16*32 = 512
    gemm_bf16_nt_wmma<<<dim3(blocksQ), blk, 0, stream>>>(xb, wqb, Qf,
                                                         (int)TOK, DIM_, DIM_);
    int blocksKV = (int)(TOK / 256) * (KVD_ / 64);  // 16*8 = 128
    gemm_bf16_nt_wmma<<<dim3(blocksKV), blk, 0, stream>>>(xb, wkb, Kf,
                                                          (int)TOK, KVD_, DIM_);
    gemm_bf16_nt_wmma<<<dim3(blocksKV), blk, 0, stream>>>(xb, wvb, Vf,
                                                          (int)TOK, KVD_, DIM_);
  }

  // 3) per-head RMSNorm + RoPE (+ gain for Q), emit bf16
  rmsnorm_rope_kernel<<<dim3((unsigned)(TOK * H_ / 8)), blk, 0, stream>>>(Qf, qbb, qg, H_);
  rmsnorm_rope_kernel<<<dim3((unsigned)(TOK * HKV_ / 8)), blk, 0, stream>>>(Kf, kbb,
                                                                            nullptr, HKV_);

  // 4) transpose V to [b][hkv][d][s] bf16 for PV fragments
  v_transpose_bf16_kernel<<<dim3((unsigned)(TOK * KVD_ / 256)), blk, 0, stream>>>(Vf, vtb);

  // 5) causal flash attention (WMMA QK^T + WMMA PV, online softmax)
  {
    int waves = B_ * H_ * (S_ / 16);                // 4096
    flash_attn_kernel<<<dim3(waves / 8), blk, 0, stream>>>(qbb, kbb, vtb, yf);
  }

  // 6) orthogonalize against normalized V direction, emit bf16
  vproj_ortho_kernel<<<dim3((unsigned)(TOK * HKV_ / 8)), blk, 0, stream>>>(yf, Vf, ybf);

  // 7) output projection (WMMA GEMM) -> f32 out
  {
    int blocks = (int)(TOK / 256) * (DIM_ / 64);    // 512
    gemm_bf16_nt_wmma<<<dim3(blocks), blk, 0, stream>>>(ybf, wob, out,
                                                        (int)TOK, DIM_, DIM_);
  }
}
